// VGDModel_773094113412
// MI455X (gfx1250) — matrix operand order repaired and checked by reading, not compiled, not run on hardware
//
#include <hip/hip_runtime.h>
#include <hip/hip_bf16.h>
#include <math.h>

// ---------------------------------------------------------------------------
// CDNA5 (gfx1250) wave32 WMMA: v_wmma_f32_16x16x32_bf16
// ---------------------------------------------------------------------------
typedef __attribute__((ext_vector_type(16))) __bf16 v16bf;
typedef __attribute__((ext_vector_type(8)))  float  v8f;

#define NB    256     // batch
#define NT    16      // timesteps
#define HIDC  128     // hidden channels per LSTM layer
#define NCOL  6400    // NB * 25 output pixels (conv GEMM N)
#define K0P   1280    // layer0 K = 140*9=1260 padded to 32
#define K1P   2304    // layer1 K = 256*9

// ---- gfx1250 async global->LDS copy (ASYNCcnt-tracked) ---------------------
static __device__ __forceinline__ void async_b128(unsigned ldsoff, const void* gptr) {
  asm volatile("global_load_async_to_lds_b128 %0, %1, off"
               :: "v"(ldsoff), "v"((unsigned long long)(uintptr_t)gptr) : "memory");
}
static __device__ __forceinline__ void wait_async0() {
  asm volatile("s_wait_asynccnt 0x0" ::: "memory");
}

// ---------------------------------------------------------------------------
// Conv GEMM (implicit im2col, B pre-materialized as [N][K] bf16):
//   D[M=512][N] = Asw[M][K] * B[N][K]^T + bias
// Asw is stored pre-swizzled in per-lane fragment order so a fragment is one
// contiguous 32B chunk: layout  [m][kb_block][g][e]  (g = lane>>4, e = 0..15).
// Block: 256 thr = 8 waves; wave tile 16x64 (4 acc, A reuse); block 64x128.
// A slab (64 rows x 32 k = 4KB) staged into LDS via async copy, double-buffered.
// ---------------------------------------------------------------------------
__global__ __launch_bounds__(256) void wmma_gemm_conv(
    const __bf16* __restrict__ Asw, const __bf16* __restrict__ Bm,
    const float* __restrict__ bias, float* __restrict__ D, int K, int N) {
  __shared__ __align__(32) __bf16 sA[2][64 * 32];   // [buf][m_local*32 + g*16 + e]
  int tid = threadIdx.x, lane = tid & 31, wave = tid >> 5;
  int wm = wave & 3, g = lane >> 4, nl = lane & 15;
  int m0 = blockIdx.y * 64 + wm * 16;
  int nb = blockIdx.x * 128 + (wave >> 2) * 64;
  int nslab = K >> 5;
  // cooperative copy role: each thread moves 16B of the 4KB slab
  int mloc = tid >> 2, piece = tid & 3;
  size_t srcrow = (size_t)(blockIdx.y * 64 + mloc) * K + piece * 8;
  unsigned dst0 = (unsigned)(uintptr_t)&sA[0][mloc * 32 + piece * 8];
  unsigned dst1 = (unsigned)(uintptr_t)&sA[1][mloc * 32 + piece * 8];

  async_b128(dst0, Asw + srcrow);           // prologue: slab 0 -> buf 0
  wait_async0();
  __syncthreads();

  v8f acc[4];
#pragma unroll
  for (int j = 0; j < 4; ++j) acc[j] = (v8f){0.f, 0.f, 0.f, 0.f, 0.f, 0.f, 0.f, 0.f};

  for (int kbi = 0; kbi < nslab; ++kbi) {
    int cur = kbi & 1;
    if (kbi + 1 < nslab)                     // prefetch next slab into other buf
      async_b128(cur ? dst0 : dst1, Asw + srcrow + (size_t)(kbi + 1) * 32);
    v16bf a = *(const v16bf*)&sA[cur][(wm * 16 + nl) * 32 + g * 16];
#pragma unroll
    for (int j = 0; j < 4; ++j) {
      v16bf b = *(const v16bf*)(Bm + (size_t)(nb + j * 16 + nl) * K + kbi * 32 + g * 16);
      acc[j] = __builtin_amdgcn_wmma_f32_16x16x32_bf16(false, a, false, b,
                                                       (short)0, acc[j], false, false);
    }
    wait_async0();
    __syncthreads();
  }
#pragma unroll
  for (int j = 0; j < 4; ++j) {
    int n = nb + j * 16 + nl;
#pragma unroll
    for (int r = 0; r < 8; ++r) {
      int m = m0 + g * 8 + r;
      D[(size_t)m * N + n] = acc[j][r] + bias[m];
    }
  }
}

// ---------------------------------------------------------------------------
// FC GEMM: D = Asw[M][K] * B[N][K]^T + bias, fused ReLU; outputs to
// f32 [M][N] and/or bf16 fused matrix [N][dbf_ld] at column offset row_off.
// ---------------------------------------------------------------------------
__global__ __launch_bounds__(256) void wmma_gemm_fc(
    const __bf16* __restrict__ Asw, const __bf16* __restrict__ Bm,
    const float* __restrict__ bias, float* __restrict__ Df32,
    __bf16* __restrict__ Dbf, int dbf_ld, int N, int K, int row_off, int relu) {
  int tid = threadIdx.x, lane = tid & 31, wave = tid >> 5;
  int wm = wave & 3, g = lane >> 4, nl = lane & 15;
  int m0 = blockIdx.y * 64 + wm * 16;
  int nb = blockIdx.x * 128 + (wave >> 2) * 64;
  int nslab = K >> 5;
  const __bf16* arow = Asw + (size_t)(m0 + nl) * K + g * 16;
  v8f acc[4];
#pragma unroll
  for (int j = 0; j < 4; ++j) acc[j] = (v8f){0.f, 0.f, 0.f, 0.f, 0.f, 0.f, 0.f, 0.f};

  for (int kbi = 0; kbi < nslab; ++kbi) {
    if (kbi + 1 < nslab)
      __builtin_prefetch(arow + (kbi + 1) * 32, 0, 1);   // global_prefetch_b8
    v16bf a = *(const v16bf*)(arow + kbi * 32);
#pragma unroll
    for (int j = 0; j < 4; ++j) {
      v16bf b = *(const v16bf*)(Bm + (size_t)(nb + j * 16 + nl) * K + kbi * 32 + g * 16);
      acc[j] = __builtin_amdgcn_wmma_f32_16x16x32_bf16(false, a, false, b,
                                                       (short)0, acc[j], false, false);
    }
  }
#pragma unroll
  for (int j = 0; j < 4; ++j) {
    int n = nb + j * 16 + nl;
#pragma unroll
    for (int r = 0; r < 8; ++r) {
      int m = m0 + g * 8 + r;
      float v = acc[j][r] + bias[m];
      if (relu) v = fmaxf(v, 0.f);
      if (Df32) Df32[(size_t)m * N + n] = v;
      if (Dbf)  Dbf[(size_t)n * dbf_ld + (m + row_off)] = (__bf16)v;
    }
  }
}

// ---------------------------------------------------------------------------
// LSTM pointwise cell. Gates z[4*128][6400] (i,f,o,g blocks), c [128][6400].
// h is scattered directly into the consumers' [N][K] im2col B-buffers:
//   Bself: own layer's conv input for t+1 (channel offset selfOff)
//   Bnext: next layer's conv input for t   (channel offset nextOff)
//   Bfc  : [256][3200] transposed FC input (final step only)
// Each (value, tap) target is unique -> race-free. Padding taps stay zero.
// ---------------------------------------------------------------------------
__global__ void lstm_pw(const float* __restrict__ z, float* __restrict__ c,
                        __bf16* __restrict__ Bself, int selfK, int selfOff,
                        __bf16* __restrict__ Bnext, int nextK, int nextOff,
                        __bf16* __restrict__ Bfc) {
  int idx = blockIdx.x * blockDim.x + threadIdx.x;
  if (idx >= HIDC * NCOL) return;
  int ch = idx / NCOL, n = idx % NCOL;
  float zi = z[(size_t)ch * NCOL + n];
  float zf = z[(size_t)(ch + HIDC) * NCOL + n];
  float zo = z[(size_t)(ch + 2 * HIDC) * NCOL + n];
  float zg = z[(size_t)(ch + 3 * HIDC) * NCOL + n];
  float si = 1.f / (1.f + __expf(-zi));
  float sf = 1.f / (1.f + __expf(-zf));
  float so = 1.f / (1.f + __expf(-zo));
  float cn = sf * c[idx] + si * tanhf(zg);
  float h  = so * tanhf(cn);
  c[idx] = cn;
  int b = n / 25, pix = n % 25, py = pix / 5, px = pix % 5;
  __bf16 hb = (__bf16)h;
#pragma unroll
  for (int ky = 0; ky < 3; ++ky) {
    int oy = py - ky + 1; if ((unsigned)oy >= 5u) continue;
#pragma unroll
    for (int kx = 0; kx < 3; ++kx) {
      int ox = px - kx + 1; if ((unsigned)ox >= 5u) continue;
      int nn = b * 25 + oy * 5 + ox, tap = ky * 3 + kx;
      Bself[(size_t)nn * selfK + (selfOff + ch) * 9 + tap] = hb;
      if (Bnext) Bnext[(size_t)nn * nextK + (nextOff + ch) * 9 + tap] = hb;
    }
  }
  if (Bfc) Bfc[(size_t)b * 3200 + ch * 25 + pix] = hb;
}

// Dynamic input (7 continuous + 5 month-emb channels) -> layer0 B-buffer taps.
__global__ void build_dyn_x(const float* __restrict__ cdyn, const int* __restrict__ catdyn,
                            const float* __restrict__ embm, __bf16* __restrict__ B0, int t) {
  int idx = blockIdx.x * blockDim.x + threadIdx.x;
  if (idx >= NB * 12 * 25) return;
  int b = idx / 300, r = idx % 300, c = r / 25, pix = r % 25;
  int py = pix / 5, px = pix % 5;
  float v;
  if (c < 7) {
    v = cdyn[(((size_t)b * 7 + c) * NT + t) * 25 + pix];
  } else {
    int m = catdyn[((size_t)b * NT + t) * 25 + pix];
    v = embm[m * 5 + (c - 7)];
  }
  __bf16 hb = (__bf16)v;
#pragma unroll
  for (int ky = 0; ky < 3; ++ky) {
    int oy = py - ky + 1; if ((unsigned)oy >= 5u) continue;
#pragma unroll
    for (int kx = 0; kx < 3; ++kx) {
      int ox = px - kx + 1; if ((unsigned)ox >= 5u) continue;
      B0[(size_t)(b * 25 + oy * 5 + ox) * K0P + c * 9 + ky * 3 + kx] = hb;
    }
  }
}

// Weight prep -> bf16 A in pre-swizzled per-lane fragment order:
//   dst[m*Kp + kb*32 + g*16 + e],  k = kb*32 + g*16 + e
// Cin>0: conv weights w[m][Cin][3][3] flattened as k=(c*9+tap); else dense [M][Ksrc].
__global__ void w_to_Asw(const float* __restrict__ w, __bf16* __restrict__ A,
                         int M, int Cin, int Ksrc, int Kp) {
  int idx = blockIdx.x * blockDim.x + threadIdx.x;
  if (idx >= M * Kp) return;
  int m = idx / Kp, tloc = idx % Kp;
  int kb = tloc >> 5, tl = tloc & 31, g = tl >> 4, e = tl & 15;
  int r = e >> 1, p = e & 1;
  int k = kb * 32 + (g << 4) + e; (void)r; (void)p;
  float v = 0.f;
  if (Cin > 0) {
    if (k < Cin * 9) v = w[((size_t)m * Cin + k / 9) * 9 + (k % 9)];
  } else {
    if (k < Ksrc) v = w[(size_t)m * Ksrc + k];
  }
  A[idx] = (__bf16)v;
}

__global__ void zero_u32(unsigned* __restrict__ p, long long n) {
  long long i = (long long)blockIdx.x * blockDim.x + threadIdx.x;
  if (i < n) p[i] = 0u;
}

// ------------------------- static branch (tiny, VALU) -----------------------
__global__ void build_static(const float* __restrict__ cstat, const int* __restrict__ catstat,
                             const float* __restrict__ eg, const float* __restrict__ el,
                             float* __restrict__ xs) {
  int idx = blockIdx.x * blockDim.x + threadIdx.x;
  if (idx >= NB * 21 * 25) return;
  int b = idx / 525, r = idx % 525, c = r / 25, pix = r % 25;
  float v;
  if (c < 8)       v = cstat[((size_t)b * 8 + c) * 25 + pix];
  else if (c < 15) v = eg[catstat[((size_t)b * 2 + 0) * 25 + pix] * 7 + (c - 8)];
  else             v = el[catstat[((size_t)b * 2 + 1) * 25 + pix] * 6 + (c - 15)];
  xs[idx] = v;
}

__global__ void static_conv1(const float* __restrict__ xs, const float* __restrict__ w,
                             const float* __restrict__ bias, const float* __restrict__ bng,
                             const float* __restrict__ bnb, float* __restrict__ out) {
  int idx = blockIdx.x * blockDim.x + threadIdx.x;
  if (idx >= NB * 16 * 49) return;
  int b = idx / (16 * 49), r = idx % (16 * 49), oc = r / 49, p = r % 49;
  int py = p / 7, px = p % 7;
  float s = bias[oc];
  for (int c = 0; c < 21; ++c)
    for (int ky = 0; ky < 3; ++ky) {
      int iy = py + ky - 2; if (iy < 0 || iy >= 5) continue;
      for (int kx = 0; kx < 3; ++kx) {
        int ix = px + kx - 2; if (ix < 0 || ix >= 5) continue;
        s += xs[((size_t)b * 21 + c) * 25 + iy * 5 + ix] *
             w[(((size_t)oc * 21 + c) * 3 + ky) * 3 + kx];
      }
    }
  s = s * (bng[oc] * rsqrtf(1.f + 1e-5f)) + bnb[oc];
  out[idx] = fmaxf(s, 0.f);
}

__global__ void pool2s1(const float* __restrict__ in, float* __restrict__ out,
                        int C, int H, int W, int total) {
  int idx = blockIdx.x * blockDim.x + threadIdx.x;
  if (idx >= total) return;
  int Ho = H - 1, Wo = W - 1;
  int b = idx / (C * Ho * Wo), r = idx % (C * Ho * Wo);
  int c = r / (Ho * Wo), p = r % (Ho * Wo), y = p / Wo, x = p % Wo;
  const float* base = in + ((size_t)b * C + c) * H * W;
  out[idx] = fmaxf(fmaxf(base[y * W + x], base[y * W + x + 1]),
                   fmaxf(base[(y + 1) * W + x], base[(y + 1) * W + x + 1]));
}

__global__ void static_conv2(const float* __restrict__ in, const float* __restrict__ w,
                             const float* __restrict__ bias, const float* __restrict__ bng,
                             const float* __restrict__ bnb, float* __restrict__ out) {
  int idx = blockIdx.x * blockDim.x + threadIdx.x;
  if (idx >= NB * 32 * 36) return;
  int b = idx / (32 * 36), r = idx % (32 * 36), oc = r / 36, p = r % 36;
  int py = p / 6, px = p % 6;
  float s = bias[oc];
  for (int c = 0; c < 16; ++c)
    for (int ky = 0; ky < 3; ++ky) {
      int iy = py + ky - 1; if (iy < 0 || iy >= 6) continue;
      for (int kx = 0; kx < 3; ++kx) {
        int ix = px + kx - 1; if (ix < 0 || ix >= 6) continue;
        s += in[((size_t)b * 16 + c) * 36 + iy * 6 + ix] *
             w[(((size_t)oc * 16 + c) * 3 + ky) * 3 + kx];
      }
    }
  s = s * (bng[oc] * rsqrtf(1.f + 1e-5f)) + bnb[oc];
  out[idx] = fmaxf(s, 0.f);
}

// final pool -> [256][800] bf16 FC input (row-major per batch)
__global__ void pool2s1_to_bstat(const float* __restrict__ in, __bf16* __restrict__ out) {
  int idx = blockIdx.x * blockDim.x + threadIdx.x;
  if (idx >= NB * 32 * 25) return;
  int b = idx / 800, r = idx % 800, oc = r / 25, p = r % 25, y = p / 5, x = p % 5;
  const float* base = in + ((size_t)b * 32 + oc) * 36;
  float m = fmaxf(fmaxf(base[y * 6 + x], base[y * 6 + x + 1]),
                  fmaxf(base[(y + 1) * 6 + x], base[(y + 1) * 6 + x + 1]));
  out[(size_t)b * 800 + oc * 25 + p] = (__bf16)m;
}

// final head: out[b] = ff2_w . h[:,b] + ff2_b   (h is [128][256] f32)
__global__ void ff2_head(const float* __restrict__ h, const float* __restrict__ w,
                         const float* __restrict__ bias, float* __restrict__ out) {
  int b = blockIdx.x * blockDim.x + threadIdx.x;
  if (b >= NB) return;
  float s = bias[0];
  for (int k = 0; k < 128; ++k) s += h[(size_t)k * NB + b] * w[k];
  out[b] = s;
}

// ---------------------------------------------------------------------------
// Host side
// ---------------------------------------------------------------------------
static inline char* wsalloc(char** p, size_t bytes) {
  char* r = *p;
  *p += (bytes + 255) & ~(size_t)255;
  return r;
}

extern "C" void kernel_launch(void* const* d_in, const int* in_sizes, int n_in,
                              void* d_out, int out_size, void* d_ws, size_t ws_size,
                              hipStream_t stream) {
  (void)in_sizes; (void)n_in; (void)out_size; (void)ws_size;
  const float* cdyn     = (const float*)d_in[0];
  const float* cstat    = (const float*)d_in[1];
  const int*   catdyn   = (const int*)d_in[2];
  const int*   catstat  = (const int*)d_in[3];
  const float* embm     = (const float*)d_in[4];
  const float* embg     = (const float*)d_in[5];
  const float* embl     = (const float*)d_in[6];
  const float* conv1_w  = (const float*)d_in[7];
  const float* conv1_b  = (const float*)d_in[8];
  const float* bn1_g    = (const float*)d_in[9];
  const float* bn1_b    = (const float*)d_in[10];
  const float* conv2_w  = (const float*)d_in[11];
  const float* conv2_b  = (const float*)d_in[12];
  const float* bn2_g    = (const float*)d_in[13];
  const float* bn2_b    = (const float*)d_in[14];
  const float* lstm0_w  = (const float*)d_in[15];
  const float* lstm0_b  = (const float*)d_in[16];
  const float* lstm1_w  = (const float*)d_in[17];
  const float* lstm1_b  = (const float*)d_in[18];
  const float* fcd_w    = (const float*)d_in[19];
  const float* fcd_b    = (const float*)d_in[20];
  const float* fcs_w    = (const float*)d_in[21];
  const float* fcs_b    = (const float*)d_in[22];
  const float* ff1_w    = (const float*)d_in[23];
  const float* ff1_b    = (const float*)d_in[24];
  const float* ff2_w    = (const float*)d_in[25];
  const float* ff2_b    = (const float*)d_in[26];
  float* out = (float*)d_out;

  // ---- workspace layout ----
  char* p = (char*)d_ws;
  __bf16* A0     = (__bf16*)wsalloc(&p, (size_t)512 * K0P * 2);
  __bf16* A1     = (__bf16*)wsalloc(&p, (size_t)512 * K1P * 2);
  __bf16* Afcd   = (__bf16*)wsalloc(&p, (size_t)256 * 3200 * 2);
  __bf16* Afcs   = (__bf16*)wsalloc(&p, (size_t)256 * 800 * 2);
  __bf16* Aff1   = (__bf16*)wsalloc(&p, (size_t)128 * 512 * 2);
  __bf16* Bc0    = (__bf16*)wsalloc(&p, (size_t)NCOL * K0P * 2);   // [6400][1280]
  __bf16* Bc1    = (__bf16*)wsalloc(&p, (size_t)NCOL * K1P * 2);   // [6400][2304]
  float*  z      = (float*) wsalloc(&p, (size_t)512 * NCOL * 4);
  float*  c0     = (float*) wsalloc(&p, (size_t)HIDC * NCOL * 4);
  float*  c1     = (float*) wsalloc(&p, (size_t)HIDC * NCOL * 4);
  __bf16* Bfc    = (__bf16*)wsalloc(&p, (size_t)NB * 3200 * 2);    // [256][3200]
  __bf16* Bstat  = (__bf16*)wsalloc(&p, (size_t)NB * 800 * 2);     // [256][800]
  __bf16* fused  = (__bf16*)wsalloc(&p, (size_t)NB * 512 * 2);     // [256][512]
  float*  ff1out = (float*) wsalloc(&p, (size_t)128 * NB * 4);
  float*  xs     = (float*) wsalloc(&p, (size_t)NB * 21 * 25 * 4);
  float*  tmp1   = (float*) wsalloc(&p, (size_t)NB * 16 * 49 * 4);
  float*  tmp2   = (float*) wsalloc(&p, (size_t)NB * 16 * 36 * 4);
  float*  tmp3   = (float*) wsalloc(&p, (size_t)NB * 32 * 36 * 4);

  const int TPB = 256;
  auto blk = [](long long n) { return (unsigned)((n + 255) / 256); };

  // ---- zero im2col buffers (padding taps + h0) and cell state ----
  zero_u32<<<blk((long long)NCOL * K0P / 2), TPB, 0, stream>>>((unsigned*)Bc0, (long long)NCOL * K0P / 2);
  zero_u32<<<blk((long long)NCOL * K1P / 2), TPB, 0, stream>>>((unsigned*)Bc1, (long long)NCOL * K1P / 2);
  zero_u32<<<blk((long long)HIDC * NCOL), TPB, 0, stream>>>((unsigned*)c0, (long long)HIDC * NCOL);
  zero_u32<<<blk((long long)HIDC * NCOL), TPB, 0, stream>>>((unsigned*)c1, (long long)HIDC * NCOL);

  // ---- weight prep (fp32 -> pre-swizzled bf16 fragment layout) ----
  w_to_Asw<<<blk(512 * K0P), TPB, 0, stream>>>(lstm0_w, A0, 512, 140, 0, K0P);
  w_to_Asw<<<blk(512 * K1P), TPB, 0, stream>>>(lstm1_w, A1, 512, 256, 0, K1P);
  w_to_Asw<<<blk(256 * 3200), TPB, 0, stream>>>(fcd_w, Afcd, 256, 0, 3200, 3200);
  w_to_Asw<<<blk(256 * 800),  TPB, 0, stream>>>(fcs_w, Afcs, 256, 0, 800, 800);
  w_to_Asw<<<blk(128 * 512),  TPB, 0, stream>>>(ff1_w, Aff1, 128, 0, 512, 512);

  // ---- static branch ----
  build_static<<<blk(NB * 21 * 25), TPB, 0, stream>>>(cstat, catstat, embg, embl, xs);
  static_conv1<<<blk(NB * 16 * 49), TPB, 0, stream>>>(xs, conv1_w, conv1_b, bn1_g, bn1_b, tmp1);
  pool2s1<<<blk(NB * 16 * 36), TPB, 0, stream>>>(tmp1, tmp2, 16, 7, 7, NB * 16 * 36);
  static_conv2<<<blk(NB * 32 * 36), TPB, 0, stream>>>(tmp2, conv2_w, conv2_b, bn2_g, bn2_b, tmp3);
  pool2s1_to_bstat<<<blk(NB * 32 * 25), TPB, 0, stream>>>(tmp3, Bstat);
  // stat_out = relu(fcs) -> fused[:, 256:512)
  wmma_gemm_fc<<<dim3(2, 4), TPB, 0, stream>>>(Afcs, Bstat, fcs_b, nullptr, fused, 512,
                                               NB, 800, 256, 1);

  // ---- ConvLSTM over time (WMMA implicit-GEMM convs) ----
  for (int t = 0; t < NT; ++t) {
    build_dyn_x<<<blk(NB * 12 * 25), TPB, 0, stream>>>(cdyn, catdyn, embm, Bc0, t);
    wmma_gemm_conv<<<dim3(NCOL / 128, 8), TPB, 0, stream>>>(A0, Bc0, lstm0_b, z, K0P, NCOL);
    lstm_pw<<<blk((long long)HIDC * NCOL), TPB, 0, stream>>>(z, c0,
        Bc0, K0P, 12, Bc1, K1P, 0, nullptr);
    wmma_gemm_conv<<<dim3(NCOL / 128, 8), TPB, 0, stream>>>(A1, Bc1, lstm1_b, z, K1P, NCOL);
    lstm_pw<<<blk((long long)HIDC * NCOL), TPB, 0, stream>>>(z, c1,
        Bc1, K1P, 128, nullptr, 0, 0, (t == NT - 1) ? Bfc : nullptr);
  }

  // ---- heads ----
  // dyn_out = relu(fcd) -> fused[:, 0:256)
  wmma_gemm_fc<<<dim3(2, 4), TPB, 0, stream>>>(Afcd, Bfc, fcd_b, nullptr, fused, 512,
                                               NB, 3200, 0, 1);
  // h = relu(ff1) -> ff1out [128][256] f32
  wmma_gemm_fc<<<dim3(2, 2), TPB, 0, stream>>>(Aff1, fused, ff1_b, ff1out, nullptr, 0,
                                               NB, 512, 0, 1);
  ff2_head<<<1, TPB, 0, stream>>>(ff1out, ff2_w, ff2_b, out);
}
